// ModelWithAuxiliaryHead_7473243095654
// MI455X (gfx1250) — compile-verified, matrix operand order checked by reading, not run on hardware
//
#include <hip/hip_runtime.h>
#include <hip/hip_bf16.h>

// ---------------- problem constants (match reference) ----------------
#define BB    2
#define SS    2048
#define HH    2048
#define VV    32000
#define NSEG  16
#define RR    256
#define TMM   1024
#define CHUNK (TMM / NSEG)   // 64

// LDS A-panel row stride: 2048 elems + 4 pads of 8 elems (16B per 512 elems,
// inserted by TDM pad_interval=256DW / pad_amount=4DW) = 2080 elems = 4160 B.
#define AROW_ELEMS 2080

typedef __attribute__((ext_vector_type(16))) __bf16         v16bf;
typedef __attribute__((ext_vector_type(8)))  float          v8f;
typedef __attribute__((ext_vector_type(8)))  unsigned short v8us;
typedef __attribute__((ext_vector_type(4)))  unsigned int   v4u;
typedef __attribute__((ext_vector_type(8)))  int            v8i;
typedef __attribute__((ext_vector_type(4)))  int            v4i;

union Frag {
    v16bf bf;
    v8us  us[2];
};

__device__ inline unsigned short f2bf(float f) {
    union { float f; unsigned u; } cv;
    cv.f = f;
    unsigned u = cv.u;
    unsigned r = u + 0x7FFFu + ((u >> 16) & 1u);   // round-to-nearest-even
    return (unsigned short)(r >> 16);
}

// ---------------- fp32 -> bf16 bulk convert ----------------
__global__ void convert_f32_bf16(const float* __restrict__ in,
                                 unsigned short* __restrict__ out, long n) {
    long i      = (long)blockIdx.x * blockDim.x + threadIdx.x;
    long stride = (long)gridDim.x * blockDim.x;
    for (; i < n; i += stride) out[i] = f2bf(in[i]);
}

// ---------------- low-rank adapter: trans = A[seg] @ (B[seg] @ x) + bias ----------------
__global__ __launch_bounds__(256) void adapter_kernel(
    const float* __restrict__ lhs,     // [B,S,H]
    const int*   __restrict__ starts,  // [B]
    const float* __restrict__ Am,      // [NSEG,H,R]
    const float* __restrict__ Bm,      // [NSEG,R,H]
    const float* __restrict__ bias,    // [NSEG,H]
    unsigned short* __restrict__ transb) // [B*TM, H] bf16
{
    __shared__ float x_s[HH];
    __shared__ float inter_s[RR];
    int token = blockIdx.x;            // 0 .. B*TM-1
    int b   = token / TMM;
    int p   = token % TMM;
    int seg = p / CHUNK;
    int pos = starts[b] + p;           // < S by construction
    const float* x = lhs + ((size_t)b * SS + pos) * HH;

    for (int h = threadIdx.x; h < HH; h += 256) x_s[h] = x[h];
    __syncthreads();

    {   // inter[r] = B[seg,r,:] . x   (one r per thread, R==256)
        int r = threadIdx.x;
        const float* brow = Bm + ((size_t)seg * RR + r) * HH;
        float acc = 0.f;
        for (int h = 0; h < HH; ++h) acc += brow[h] * x_s[h];
        inter_s[r] = acc;
    }
    __syncthreads();

    for (int h = threadIdx.x; h < HH; h += 256) {
        const float* arow = Am + ((size_t)seg * HH + h) * RR;
        float acc = bias[seg * HH + h];
        for (int r = 0; r < RR; ++r) acc += arow[r] * inter_s[r];
        transb[(size_t)token * HH + h] = f2bf(acc);
    }
}

// ---------------- WMMA online-softmax vocab GEMM ----------------
// Grid: (ceil(rowsPerBatch/16), B). A block owns 16 token rows of one batch,
// staged once into LDS by the Tensor Data Mover (zero-filled OOB tail rows,
// 16B pad per 512 elems for bank spread + alignment). The 8 waves split the
// vocab (disjoint 16-col tiles), so W is read once per block. During the
// sweep every lane keeps private (m, sumexp, label) stats for the disjoint
// column subset it sees -> no shuffles in the hot loop. Lanes are merged with
// one butterfly pass at the end, then the 8 waves merge via LDS.
__global__ __launch_bounds__(256) void gemm_softmax_nll(
    const unsigned short* __restrict__ X,   // bf16 activations
    const unsigned short* __restrict__ Wb,  // [V,H] bf16
    const int* __restrict__ labels, int labelStride, int labelOffset,
    int rowsPerBatch, int xRowStride,
    float* __restrict__ nll_out)            // [B*rowsPerBatch]
{
    __shared__ __align__(128) unsigned short a_s[16 * AROW_ELEMS]; // 66,560 B
    __shared__ float m_s[8][16];
    __shared__ float l_s[8][16];
    __shared__ float g_s[8][16];
    __shared__ int   lab_sh[16];

    int tid   = threadIdx.x;
    int lane  = tid & 31;
    int wave  = tid >> 5;
    int batch = blockIdx.y;
    int p0    = blockIdx.x * 16;

    // ---- stage A panel via TDM (wave 0 only; EXEC ignored by TDM) ----
    if (tid < 32) {
        int nrows = rowsPerBatch - p0;
        if (nrows > 16) nrows = 16;
        unsigned long long ga =
            (unsigned long long)(uintptr_t)(X + ((size_t)batch * xRowStride + p0) * HH);
        unsigned ldsb = (unsigned)(uintptr_t)&a_s[0];   // low 32 bits = LDS byte addr
        v4u g0 = { 1u,                                  // count=1
                   ldsb,                                // lds_addr
                   (unsigned)(ga & 0xFFFFFFFFu),        // global_addr[31:0]
                   (unsigned)((ga >> 32) & 0x01FFFFFFu) | 0x80000000u }; // [56:32] | type=2
        v8i g1 = { (int)0x07D10000,      // data_size=2B | pad_en | interval=256DW | amount=4DW
                   (int)0x08000000,      // tensor_dim0 = 2048 (<<16)
                   nrows << 16,          // tensor_dim1 = valid rows (OOB rows -> 0)
                   (int)0x08000000,      // tile_dim0 = 2048 (<<16)
                   16,                   // tile_dim1 = 16
                   2048,                 // tensor_dim0_stride = H
                   0, 0 };
        v4i gz4 = { 0, 0, 0, 0 };
        v8i gz8 = { 0, 0, 0, 0, 0, 0, 0, 0 };
        __builtin_amdgcn_tensor_load_to_lds(g0, g1, gz4, gz4, gz8, 0);
        __builtin_amdgcn_s_wait_tensorcnt(0);
    }
    if (tid < 16) {
        int p  = p0 + tid;
        int pc = p < rowsPerBatch ? p : rowsPerBatch - 1;
        lab_sh[tid] = labels[batch * labelStride + pc + labelOffset];
    }
    __syncthreads();

    const int kbase = (lane < 16) ? 0 : 8;   // bf16 A/B fragment K split
    const int hi    = (lane >= 16) ? 8 : 0;  // C layout: lanes 16-31 hold M=i+8
    const int aoff  = (lane & 15) * AROW_ELEMS;

    float mrow[8], lrow[8], labv[8];
    int   myLab[8];
#pragma unroll
    for (int i = 0; i < 8; ++i) {
        mrow[i] = -3.0e38f; lrow[i] = 0.f; labv[i] = 0.f;
        myLab[i] = lab_sh[i + hi];
    }

    for (int t = 0; t < VV / (16 * 8); ++t) {          // 250 tiles per wave
        int col = (t * 8 + wave) * 16 + (lane & 15);
        const unsigned short* wrow = Wb + (size_t)col * HH;
        v8f c = {};
        for (int kk = 0; kk < HH; kk += 32) {
            int e0 = aoff + kk + kbase + ((kk >> 9) << 3);  // +8 elems pad / 512
            Frag a, b;
            a.us[0] = *(const v8us*)&a_s[e0];
            a.us[1] = *(const v8us*)&a_s[e0 + 16];
            b.us[0] = *(const v8us*)(wrow + kk + kbase);
            b.us[1] = *(const v8us*)(wrow + kk + kbase + 16);
            c = __builtin_amdgcn_wmma_f32_16x16x32_bf16(
                    false, a.bf, false, b.bf, (short)0, c, false, false);
        }
        // lane-local online softmax: each lane owns a disjoint column subset
#pragma unroll
        for (int i = 0; i < 8; ++i) {
            float v  = c[i];
            float nm = fmaxf(mrow[i], v);
            lrow[i]  = lrow[i] * __expf(mrow[i] - nm) + __expf(v - nm);
            mrow[i]  = nm;
            labv[i] += (col == myLab[i]) ? v : 0.f;
        }
    }

    // ---- merge the 16 lanes of each half (butterfly, once) ----
#pragma unroll
    for (int i = 0; i < 8; ++i) {
        float M = mrow[i], L = lrow[i], G = labv[i];
#pragma unroll
        for (int mask = 1; mask <= 8; mask <<= 1) {
            float Mo = __shfl_xor(M, mask);
            float Lo = __shfl_xor(L, mask);
            float Go = __shfl_xor(G, mask);
            float nM = fmaxf(M, Mo);
            L = L * __expf(M - nM) + Lo * __expf(Mo - nM);
            M = nM;
            G += Go;
        }
        mrow[i] = M; lrow[i] = L; labv[i] = G;
    }

    // ---- publish per-wave stats, then 8-way merge per row ----
    if ((lane & 15) == 0) {
#pragma unroll
        for (int i = 0; i < 8; ++i) {
            m_s[wave][hi + i] = mrow[i];
            l_s[wave][hi + i] = lrow[i];
            g_s[wave][hi + i] = labv[i];
        }
    }
    __syncthreads();

    if (tid < 16) {
        float M = -3.0e38f;
#pragma unroll
        for (int w = 0; w < 8; ++w) M = fmaxf(M, m_s[w][tid]);
        float L = 0.f, LAB = 0.f;
#pragma unroll
        for (int w = 0; w < 8; ++w) {
            L   += l_s[w][tid] * __expf(m_s[w][tid] - M);
            LAB += g_s[w][tid];
        }
        int p = p0 + tid;
        if (p < rowsPerBatch)
            nll_out[batch * rowsPerBatch + p] = (M + __logf(L)) - LAB;
    }
}

// ---------------- masked means + weighted total ----------------
__global__ __launch_bounds__(256) void reduce_losses(
    const float* __restrict__ nll_main,     // [B*(S-1)]
    const float* __restrict__ nll_math,     // [B*TM]
    const int* __restrict__ attention_mask, // [B,S]
    const int* __restrict__ starts,
    const int* __restrict__ ends,
    const int* __restrict__ math_mask,      // [B,TM]
    float* __restrict__ out)                // 4 floats
{
    __shared__ float red[256];
    __shared__ int   rl_s[BB];
    int tid = threadIdx.x;

    for (int b = 0; b < BB; ++b) {
        float partial = 0.f;
        for (int j = tid; j < SS; j += 256) partial += (float)attention_mask[b * SS + j];
        red[tid] = partial; __syncthreads();
        for (int s = 128; s > 0; s >>= 1) { if (tid < s) red[tid] += red[tid + s]; __syncthreads(); }
        if (tid == 0) rl_s[b] = (int)red[0];
        __syncthreads();
    }

    float acc[6] = {0.f, 0.f, 0.f, 0.f, 0.f, 0.f};
    for (int r = tid; r < BB * (SS - 1); r += 256) {
        int b = r / (SS - 1), p = r % (SS - 1);
        float v = nll_main[r];
        int stM = (p >= starts[b] - 1) && (p <= ends[b] - 1);
        int faM = (p >= ends[b]) && (p < rl_s[b] - 1);
        acc[0] += stM ? v : 0.f; acc[1] += (float)stM;
        acc[2] += faM ? v : 0.f; acc[3] += (float)faM;
    }
    for (int r = tid; r < BB * TMM; r += 256) {
        float mk = (float)math_mask[r];
        acc[4] += nll_math[r] * mk; acc[5] += mk;
    }

    float tot[6];
    for (int k = 0; k < 6; ++k) {
        red[tid] = acc[k]; __syncthreads();
        for (int s = 128; s > 0; s >>= 1) { if (tid < s) red[tid] += red[tid + s]; __syncthreads(); }
        tot[k] = red[0]; __syncthreads();
    }
    if (tid == 0) {
        float st = tot[0] / fmaxf(tot[1], 1.f);
        float fa = tot[2] / fmaxf(tot[3], 1.f);
        float ml = tot[4] / fmaxf(tot[5], 1.f);
        out[0] = 0.5f * ml + 0.5f * st + 0.4f * fa;
        out[1] = ml; out[2] = st; out[3] = fa;
    }
}

// ---------------- host launch ----------------
extern "C" void kernel_launch(void* const* d_in, const int* in_sizes, int n_in,
                              void* d_out, int out_size, void* d_ws, size_t ws_size,
                              hipStream_t stream) {
    const float* lhs        = (const float*)d_in[0];   // [B,S,H]
    const int*   input_ids  = (const int*)d_in[1];     // [B,S]
    const int*   attn_mask  = (const int*)d_in[2];     // [B,S]
    const int*   starts     = (const int*)d_in[3];     // [B]
    const int*   ends       = (const int*)d_in[4];     // [B]
    const int*   math_lab   = (const int*)d_in[5];     // [B,TM]
    const int*   math_mask  = (const int*)d_in[6];     // [B,TM]
    /* d_in[7] math_lengths (int64) unused */
    const float* Am         = (const float*)d_in[8];   // [NSEG,H,R]
    const float* Bm         = (const float*)d_in[9];   // [NSEG,R,H]
    const float* bias       = (const float*)d_in[10];  // [NSEG,H]
    const float* W          = (const float*)d_in[11];  // [V,H]
    float* out              = (float*)d_out;

    char* ws = (char*)d_ws;
    size_t off = 0;
    unsigned short* Wb     = (unsigned short*)(ws + off); off += (size_t)VV * HH * 2;
    unsigned short* hsb    = (unsigned short*)(ws + off); off += (size_t)BB * SS * HH * 2;
    unsigned short* transb = (unsigned short*)(ws + off); off += (size_t)BB * TMM * HH * 2;
    float* nll_main        = (float*)(ws + off);          off += (size_t)BB * (SS - 1) * 4;
    float* nll_math        = (float*)(ws + off);

    // 1) bf16 conversions (W in bf16 = 131 MB -> fits the 192 MB L2)
    convert_f32_bf16<<<4096, 256, 0, stream>>>(W,   Wb,  (long)VV * HH);
    convert_f32_bf16<<<2048, 256, 0, stream>>>(lhs, hsb, (long)BB * SS * HH);

    // 2) low-rank adapter -> bf16 adapted states
    adapter_kernel<<<BB * TMM, 256, 0, stream>>>(lhs, starts, Am, Bm, bias, transb);

    // 3) online-softmax vocab GEMMs (TDM-staged A panel, WMMA bf16, f32 accum)
    {
        int rowsPerBatch = SS - 1;                           // 2047
        int rowBlocks = (rowsPerBatch + 15) / 16;            // 128
        gemm_softmax_nll<<<dim3(rowBlocks, BB), 256, 0, stream>>>(
            hsb, Wb, input_ids, /*labelStride=*/SS, /*labelOffset=*/1,
            rowsPerBatch, /*xRowStride=*/SS, nll_main);
    }
    {
        int rowsPerBatch = TMM;                              // 1024
        int rowBlocks = (rowsPerBatch + 15) / 16;            // 64
        gemm_softmax_nll<<<dim3(rowBlocks, BB), 256, 0, stream>>>(
            transb, Wb, math_lab, /*labelStride=*/TMM, /*labelOffset=*/0,
            rowsPerBatch, /*xRowStride=*/TMM, nll_math);
    }

    // 4) masked means + weighted total -> 4 scalars
    reduce_losses<<<1, 256, 0, stream>>>(nll_main, nll_math, attn_mask,
                                         starts, ends, math_mask, out);
}
